// RnnSenderReinforce_19026705121661
// MI455X (gfx1250) — compile-verified, matrix operand order checked by reading, not run on hardware
//
#include <hip/hip_runtime.h>
#include <hip/hip_bf16.h>
#include <stdint.h>

// ---------------------------------------------------------------------------
// RnnSenderReinforce for MI455X (gfx1250, wave32).
// - bf16 WMMA (v_wmma_f32_16x16x32_bf16) GEMMs, f32 accumulation.
// - GEMM uses the CDNA5 async-copy path: GLOBAL_LOAD_ASYNC_TO_LDS_B128
//   (ASYNCcnt) double-buffered staging of A/W tiles into LDS, fragments
//   read back with ds_load_b128.
// - Fused GRU-gate kernel keeps all 6 gate accumulators (gi_r/z/n, gh_r/z/n)
//   in VGPRs so the 2x[B,3H] intermediates never touch HBM.
// ---------------------------------------------------------------------------

typedef __attribute__((ext_vector_type(16))) __bf16 bf16x16;
typedef __attribute__((ext_vector_type(8)))  __bf16 bf16x8;
typedef __attribute__((ext_vector_type(8)))  float  f32x8;

__device__ __forceinline__ f32x8 wmma_bf16f32(bf16x16 a, bf16x16 b, f32x8 c) {
  // D = A(16x32 bf16) * B(32x16 bf16) + C(16x16 f32)
  return __builtin_amdgcn_wmma_f32_16x16x32_bf16(
      /*neg_a=*/false, a, /*neg_b=*/false, b,
      /*c_mod=*/(short)0, c, /*reuse_a=*/false, /*reuse_b=*/false);
}

// Load one 16x32 bf16 WMMA operand fragment from a row-major [rows, ld]
// matrix with K contiguous. Works for A (row0 = M-tile) and for B = W^T
// (row0 = N-tile into W's rows), since B(k,n) = W[n,k].
// Per ISA 7.12.2: lane L (L<16) holds row M=L, K = {0..7, 16..23};
// lane L+16 holds row M=L, K = {8..15, 24..31}.  -> two 16B loads per lane.
__device__ __forceinline__ bf16x16 load_frag(const __bf16* __restrict__ p,
                                             int ld, int row0, int k0, int lane) {
  const int r  = row0 + (lane & 15);
  const int kb = k0 + ((lane >> 4) << 3);
  const __bf16* q = p + (size_t)r * ld + kb;
  union { bf16x16 v; bf16x8 h[2]; } f;
  f.h[0] = *(const bf16x8*)(q);
  f.h[1] = *(const bf16x8*)(q + 16);
  return f.v;
}

// CDNA5 async copy: 16B/lane global -> LDS, tracked by ASYNCcnt (ISA ch.10/15.18).
__device__ __forceinline__ void async_ld_b128(unsigned ldsOff, unsigned long long gaddr) {
  asm volatile("global_load_async_to_lds_b128 %0, %1, off"
               :: "v"(ldsOff), "v"(gaddr) : "memory");
}
__device__ __forceinline__ void wait_asynccnt3() {
  asm volatile("s_wait_asynccnt 0x3" ::: "memory");
}
__device__ __forceinline__ void wait_asynccnt0() {
  asm volatile("s_wait_asynccnt 0x0" ::: "memory");
}

// ---------------------------------------------------------------------------
// small elementwise helpers
// ---------------------------------------------------------------------------
__global__ void k_cvt_f32_bf16(const float* __restrict__ s, __bf16* __restrict__ d, int n) {
  for (int i = blockIdx.x * blockDim.x + threadIdx.x; i < n; i += gridDim.x * blockDim.x)
    d[i] = (__bf16)s[i];
}

__global__ void k_broadcast_sos(const __bf16* __restrict__ sos, __bf16* __restrict__ e, int n, int E1) {
  for (int i = blockIdx.x * blockDim.x + threadIdx.x; i < n; i += gridDim.x * blockDim.x)
    e[i] = sos[i & (E1 - 1)];
}

__global__ void k_zero_tail(int* seq, float* lp, float* ent, int Lp1, int B1) {
  int b = blockIdx.x * blockDim.x + threadIdx.x;
  if (b < B1) {
    seq[(size_t)b * Lp1 + (Lp1 - 1)] = 0;
    lp [(size_t)b * Lp1 + (Lp1 - 1)] = 0.f;
    ent[(size_t)b * Lp1 + (Lp1 - 1)] = 0.f;
  }
}

// ---------------------------------------------------------------------------
// GEMM: C[M,N] = A[M,K](bf16) @ W[N,K](bf16)^T + bias.
// Workgroup (256 thr = 8 waves, 4Mx2N) computes a 128x64 macro-tile.
// Per 32-wide K-step: each thread issues 3 GLOBAL_LOAD_ASYNC_TO_LDS_B128
// (A tile 128x32 = 8KB, W tile 64x32 = 4KB) into the next LDS buffer while
// WMMAs consume the current one; s_wait_asynccnt + barriers synchronize.
// Each wave: 32x32 tile -> 8 ds_load_b128 + 4 WMMA per K-step.
// ---------------------------------------------------------------------------
#define GT_BUF (128 * 32 + 64 * 32)  // bf16 elems per LDS buffer (12KB)

__global__ void k_gemm_bf16_lds(const __bf16* __restrict__ A,
                                const __bf16* __restrict__ W,
                                const float* __restrict__ bias,
                                float* __restrict__ Cf,
                                __bf16* __restrict__ Cb,
                                int M, int N, int K) {
  __shared__ __bf16 smem[2 * GT_BUF];

  const int tid  = threadIdx.x;
  const int lane = tid & 31;
  const int wv   = tid >> 5;          // 0..7
  const int wm   = (wv >> 1) * 32;    // wave row offset in 128-tile
  const int wn   = (wv & 1) * 32;     // wave col offset in 64-tile

  const int nBlk = N >> 6;
  const int m0 = (blockIdx.x / nBlk) * 128;
  const int n0 = (blockIdx.x % nBlk) * 64;

  const unsigned ldsBase = (unsigned)(size_t)(&smem[0]);

  // stage K-slice `k` into LDS buffer `buf` (3 async b128 per thread)
  auto stage = [&](int buf, int k) {
    const unsigned aOff = ldsBase + (unsigned)buf * (GT_BUF * 2);
    const unsigned bOff = aOff + 128 * 64;  // bytes: A tile is 128 rows * 64B
    // A tile: 128 rows x 64B, 512 chunks of 16B -> 2 per thread
    {
      const int c0 = tid * 2;
      int row = c0 >> 2, off = (c0 & 3) * 16;
      async_ld_b128(aOff + row * 64 + off,
                    (unsigned long long)(size_t)((const char*)A + ((size_t)(m0 + row) * K + k) * 2 + off));
      const int c1 = c0 + 1;
      row = c1 >> 2; off = (c1 & 3) * 16;
      async_ld_b128(aOff + row * 64 + off,
                    (unsigned long long)(size_t)((const char*)A + ((size_t)(m0 + row) * K + k) * 2 + off));
    }
    // W tile: 64 rows x 64B, 256 chunks of 16B -> 1 per thread
    {
      const int row = tid >> 2, off = (tid & 3) * 16;
      async_ld_b128(bOff + row * 64 + off,
                    (unsigned long long)(size_t)((const char*)W + ((size_t)(n0 + row) * K + k) * 2 + off));
    }
  };

  f32x8 acc00 = {}, acc01 = {}, acc10 = {}, acc11 = {};

  stage(0, 0);
  for (int k = 0; k < K; k += 32) {
    const int buf = (k >> 5) & 1;
    if (k > 0) __syncthreads();           // prior readers of buf^1 are done
    if (k + 32 < K) {
      stage(buf ^ 1, k + 32);             // overlap next stage with compute
      wait_asynccnt3();                   // this wave's previous stage landed
    } else {
      wait_asynccnt0();
    }
    __syncthreads();                      // all waves' stages landed

    const __bf16* lA = &smem[buf * GT_BUF];
    const __bf16* lB = &smem[buf * GT_BUF + 128 * 32];
    bf16x16 a0 = load_frag(lA, 32, wm,      0, lane);
    bf16x16 a1 = load_frag(lA, 32, wm + 16, 0, lane);
    bf16x16 b0 = load_frag(lB, 32, wn,      0, lane);
    bf16x16 b1 = load_frag(lB, 32, wn + 16, 0, lane);
    acc00 = wmma_bf16f32(a0, b0, acc00);
    acc01 = wmma_bf16f32(a0, b1, acc01);
    acc10 = wmma_bf16f32(a1, b0, acc10);
    acc11 = wmma_bf16f32(a1, b1, acc11);
  }

  const int nl = lane & 15;
  const int mo = (lane >> 4) << 3;  // 0 or 8
#pragma unroll
  for (int i = 0; i < 2; i++) {
#pragma unroll
    for (int j = 0; j < 2; j++) {
      const f32x8 acc = (i == 0) ? (j == 0 ? acc00 : acc01)
                                 : (j == 0 ? acc10 : acc11);
      const int n = n0 + wn + j * 16 + nl;
      const float bv = bias ? bias[n] : 0.f;
      const int mbase = m0 + wm + i * 16 + mo;
#pragma unroll
      for (int r = 0; r < 8; r++) {
        const float v = acc[r] + bv;
        const size_t idx = (size_t)(mbase + r) * N + n;
        if (Cf) Cf[idx] = v;
        if (Cb) Cb[idx] = (__bf16)v;
      }
    }
  }
}

// ---------------------------------------------------------------------------
// Fused GRU cell (torch.nn.GRUCell math).
// One wave -> 32 rows x 16 cols of H.  12 WMMA accumulators live in VGPRs:
// gi_{r,z,n} and gh_{r,z,n} for two M-subtiles. Epilogue does the gate math
// in fp32 against the fp32 master hidden state (double buffered).
// ---------------------------------------------------------------------------
__global__ void k_gru_fused(const __bf16* __restrict__ e,     // [B,E] bf16
                            const __bf16* __restrict__ hb,    // [B,H] bf16 (h_t)
                            const float*  __restrict__ hf,    // [B,H] f32  (h_t)
                            const __bf16* __restrict__ Wih,   // [3H,E]
                            const __bf16* __restrict__ Whh,   // [3H,H]
                            const float*  __restrict__ bih,   // [3H]
                            const float*  __restrict__ bhh,   // [3H]
                            float*  __restrict__ hf_out,      // [B,H] f32  (h_{t+1})
                            __bf16* __restrict__ hb_out) {    // [B,H] bf16 (h_{t+1})
  const int H = 512, E = 512;
  const int lane = threadIdx.x & 31;
  const int wave = (blockIdx.x * blockDim.x + threadIdx.x) >> 5;
  const int ntiles = H >> 4;  // 32 column tiles of 16
  const int mt = wave / ntiles, nt = wave % ntiles;
  const int m0 = mt * 32, n0 = nt * 16;

  f32x8 gi[2][3] = {{{}, {}, {}}, {{}, {}, {}}};
  f32x8 gh[2][3] = {{{}, {}, {}}, {{}, {}, {}}};

  // gi = e @ W_ih^T  (three gate column-tiles share the A fragments)
  for (int k = 0; k < E; k += 32) {
    bf16x16 a0 = load_frag(e, E, m0,      k, lane);
    bf16x16 a1 = load_frag(e, E, m0 + 16, k, lane);
    bf16x16 br = load_frag(Wih, E, n0,         k, lane);
    bf16x16 bz = load_frag(Wih, E, H + n0,     k, lane);
    bf16x16 bn = load_frag(Wih, E, 2 * H + n0, k, lane);
    gi[0][0] = wmma_bf16f32(a0, br, gi[0][0]);
    gi[0][1] = wmma_bf16f32(a0, bz, gi[0][1]);
    gi[0][2] = wmma_bf16f32(a0, bn, gi[0][2]);
    gi[1][0] = wmma_bf16f32(a1, br, gi[1][0]);
    gi[1][1] = wmma_bf16f32(a1, bz, gi[1][1]);
    gi[1][2] = wmma_bf16f32(a1, bn, gi[1][2]);
  }
  // gh = h @ W_hh^T
  for (int k = 0; k < H; k += 32) {
    bf16x16 a0 = load_frag(hb, H, m0,      k, lane);
    bf16x16 a1 = load_frag(hb, H, m0 + 16, k, lane);
    bf16x16 br = load_frag(Whh, H, n0,         k, lane);
    bf16x16 bz = load_frag(Whh, H, H + n0,     k, lane);
    bf16x16 bn = load_frag(Whh, H, 2 * H + n0, k, lane);
    gh[0][0] = wmma_bf16f32(a0, br, gh[0][0]);
    gh[0][1] = wmma_bf16f32(a0, bz, gh[0][1]);
    gh[0][2] = wmma_bf16f32(a0, bn, gh[0][2]);
    gh[1][0] = wmma_bf16f32(a1, br, gh[1][0]);
    gh[1][1] = wmma_bf16f32(a1, bz, gh[1][1]);
    gh[1][2] = wmma_bf16f32(a1, bn, gh[1][2]);
  }

  // epilogue: r = sig(ir+hr), z = sig(iz+hz), n = tanh(in + r*hn),
  //           h' = (1-z)*n + z*h
  const int n = n0 + (lane & 15);
  const int mo = (lane >> 4) << 3;
  const float bir = bih[n], biz = bih[H + n], bin_ = bih[2 * H + n];
  const float bhr = bhh[n], bhz = bhh[H + n], bhn = bhh[2 * H + n];
#pragma unroll
  for (int i = 0; i < 2; i++) {
    const int mbase = m0 + i * 16 + mo;
#pragma unroll
    for (int r = 0; r < 8; r++) {
      const float ir = gi[i][0][r] + bir, iz = gi[i][1][r] + biz, in_ = gi[i][2][r] + bin_;
      const float hr = gh[i][0][r] + bhr, hz = gh[i][1][r] + bhz, hn = gh[i][2][r] + bhn;
      const float rg = 1.f / (1.f + __expf(-(ir + hr)));
      const float zg = 1.f / (1.f + __expf(-(iz + hz)));
      const float ng = tanhf(in_ + rg * hn);
      const size_t idx = (size_t)(mbase + r) * H + n;
      const float hv = (1.f - zg) * ng + zg * hf[idx];
      hf_out[idx] = hv;
      hb_out[idx] = (__bf16)hv;
    }
  }
}

// ---------------------------------------------------------------------------
// Per-row (one wave32 per batch row): log_softmax stats, entropy,
// Gumbel-max categorical sample, output writes + embedding gather.
// ---------------------------------------------------------------------------
__device__ __forceinline__ unsigned hash3(unsigned a, unsigned b, unsigned c) {
  unsigned h = a * 0x9E3779B1u ^ b * 0x85EBCA77u ^ c * 0xC2B2AE3Du;
  h ^= h >> 16; h *= 0x7FEB352Du; h ^= h >> 15; h *= 0x846CA68Bu; h ^= h >> 16;
  return h;
}

__global__ void k_softmax_sample(const float* __restrict__ logits,  // [B,V]
                                 const __bf16* __restrict__ emb,    // [V,E] bf16
                                 __bf16* __restrict__ e_out,        // [B,E] bf16
                                 int* __restrict__ seq, float* __restrict__ lp_out,
                                 float* __restrict__ ent_out, int t, int Lp1) {
  const int V = 1024, E = 512;
  const int lane = threadIdx.x & 31;
  const int row  = (blockIdx.x * blockDim.x + threadIdx.x) >> 5;
  const float* lrow = logits + (size_t)row * V;

  float v[32];
  float mx = -3.402823466e+38f;
#pragma unroll
  for (int i = 0; i < 32; i++) {
    v[i] = lrow[lane + i * 32];
    mx = fmaxf(mx, v[i]);
  }
#pragma unroll
  for (int o = 16; o; o >>= 1) mx = fmaxf(mx, __shfl_xor(mx, o, 32));

  float s = 0.f, sl = 0.f;
  float best = -3.402823466e+38f, bval = 0.f;
  int bidx = 0;
#pragma unroll
  for (int i = 0; i < 32; i++) {
    const int col = lane + i * 32;
    const float ex = __expf(v[i] - mx);
    s += ex;
    sl += ex * v[i];
    // deterministic counter-based Gumbel noise (distributional stand-in for
    // jax.random.categorical; argmax(logits+g) == argmax(logp+g))
    const unsigned h = hash3((unsigned)row, (unsigned)col, (unsigned)(t + 1));
    const float u = (float)(h >> 8) * (1.f / 16777216.f);
    const float g = -__logf(-__logf(u + 1e-20f) + 1e-20f);
    const float cand = v[i] + g;
    if (cand > best) { best = cand; bidx = col; bval = v[i]; }
  }
#pragma unroll
  for (int o = 16; o; o >>= 1) {
    const float ob = __shfl_xor(best, o, 32);
    const int   oi = __shfl_xor(bidx, o, 32);
    const float ov = __shfl_xor(bval, o, 32);
    if (ob > best || (ob == best && oi < bidx)) { best = ob; bidx = oi; bval = ov; }
    s  += __shfl_xor(s,  o, 32);
    sl += __shfl_xor(sl, o, 32);
  }

  const float lse = mx + __logf(s);
  const float ent = lse - sl / s;     // -sum(p * logp)
  if (lane == 0) {
    seq    [(size_t)row * Lp1 + t] = bidx;
    lp_out [(size_t)row * Lp1 + t] = bval - lse;
    ent_out[(size_t)row * Lp1 + t] = ent;
  }

  // e_{t+1} = embedding[sym] : 512 bf16 = 32 lanes x 2 x 16B
  const bf16x8* src = (const bf16x8*)(emb + (size_t)bidx * E);
  bf16x8* dst = (bf16x8*)(e_out + (size_t)row * E);
  dst[lane * 2]     = src[lane * 2];
  dst[lane * 2 + 1] = src[lane * 2 + 1];
}

// ---------------------------------------------------------------------------
// host launcher
// ---------------------------------------------------------------------------
extern "C" void kernel_launch(void* const* d_in, const int* in_sizes, int n_in,
                              void* d_out, int out_size, void* d_ws, size_t ws_size,
                              hipStream_t stream) {
  (void)in_sizes; (void)n_in; (void)out_size; (void)ws_size;
  const int B = 8192, NF = 512, H = 512, E = 512, V = 1024, L = 32, Lp1 = L + 1;

  const float* x      = (const float*)d_in[0];
  const float* Wag    = (const float*)d_in[1];
  const float* bag    = (const float*)d_in[2];
  const float* Wih    = (const float*)d_in[3];
  const float* Whh    = (const float*)d_in[4];
  const float* bih    = (const float*)d_in[5];
  const float* bhh    = (const float*)d_in[6];
  const float* embF   = (const float*)d_in[7];
  const float* sosF   = (const float*)d_in[8];
  const float* Wout   = (const float*)d_in[9];
  const float* bout   = (const float*)d_in[10];

  // workspace carve-up
  char* w = (char*)d_ws;
  auto alloc = [&](size_t bytes) {
    char* p = w;
    w += (bytes + 255) & ~(size_t)255;
    return p;
  };
  __bf16* x_b    = (__bf16*)alloc((size_t)B * NF * 2);
  __bf16* Wag_b  = (__bf16*)alloc((size_t)H * NF * 2);
  __bf16* Wih_b  = (__bf16*)alloc((size_t)3 * H * E * 2);
  __bf16* Whh_b  = (__bf16*)alloc((size_t)3 * H * H * 2);
  __bf16* Wout_b = (__bf16*)alloc((size_t)V * H * 2);
  __bf16* emb_b  = (__bf16*)alloc((size_t)V * E * 2);
  __bf16* sos_b  = (__bf16*)alloc((size_t)E * 2);
  float*  hA_f   = (float*)alloc((size_t)B * H * 4);
  float*  hB_f   = (float*)alloc((size_t)B * H * 4);
  __bf16* hA_b   = (__bf16*)alloc((size_t)B * H * 2);
  __bf16* hB_b   = (__bf16*)alloc((size_t)B * H * 2);
  __bf16* e_b    = (__bf16*)alloc((size_t)B * E * 2);
  float*  logit  = (float*)alloc((size_t)B * V * 4);

  int*   seq  = (int*)d_out;
  float* lpo  = (float*)d_out + (size_t)B * Lp1;
  float* ento = (float*)d_out + (size_t)2 * B * Lp1;

  auto cvt = [&](const float* s, __bf16* d, int n) {
    int blocks = (n + 255) / 256;
    if (blocks > 4096) blocks = 4096;
    k_cvt_f32_bf16<<<blocks, 256, 0, stream>>>(s, d, n);
  };
  cvt(x, x_b, B * NF);
  cvt(Wag, Wag_b, H * NF);
  cvt(Wih, Wih_b, 3 * H * E);
  cvt(Whh, Whh_b, 3 * H * H);
  cvt(Wout, Wout_b, V * H);
  cvt(embF, emb_b, V * E);
  cvt(sosF, sos_b, E);

  // h0 = x @ W_agent^T + b_agent  -> fp32 + bf16 copies (128x64 macro-tiles)
  k_gemm_bf16_lds<<<(B / 128) * (H / 64), 256, 0, stream>>>(
      x_b, Wag_b, bag, hA_f, hA_b, B, H, NF);
  // e0 = broadcast(sos)
  k_broadcast_sos<<<4096, 256, 0, stream>>>(sos_b, e_b, B * E, E);

  float*  hc_f = hA_f;  __bf16* hc_b = hA_b;
  float*  hn_f = hB_f;  __bf16* hn_b = hB_b;

  for (int t = 0; t < L; t++) {
    // h_{t+1} = GRUCell(e_t, h_t)
    {
      int waves = (B / 32) * (H / 16);
      k_gru_fused<<<waves * 32 / 256, 256, 0, stream>>>(
          e_b, hc_b, hc_f, Wih_b, Whh_b, bih, bhh, hn_f, hn_b);
    }
    // logits = h_{t+1} @ W_out^T + b_out
    k_gemm_bf16_lds<<<(B / 128) * (V / 64), 256, 0, stream>>>(
        hn_b, Wout_b, bout, logit, (__bf16*)nullptr, B, V, H);
    // sample + stats + e_{t+1} = embedding[sym]
    k_softmax_sample<<<B * 32 / 256, 256, 0, stream>>>(
        logit, emb_b, e_b, seq, lpo, ento, t, Lp1);

    // swap double buffers
    float* tf = hc_f; hc_f = hn_f; hn_f = tf;
    __bf16* tb = hc_b; hc_b = hn_b; hn_b = tb;
  }

  // trailing zero column (t = L)
  k_zero_tail<<<(B + 255) / 256, 256, 0, stream>>>(seq, lpo, ento, Lp1, B);
}
